// TgramNet3_55130200212150
// MI455X (gfx1250) — matrix operand order repaired and checked by reading, not compiled, not run on hardware
//
#include <hip/hip_runtime.h>

// ---------------------------------------------------------------------------
// TgramNet on MI455X (gfx1250): all GEMMs via v_wmma_f32_16x16x32_bf16,
// extractor input staged to LDS with the Tensor Data Mover (TDM).
// ---------------------------------------------------------------------------

typedef __bf16 bf16;
typedef bf16  v16bf __attribute__((ext_vector_type(16)));
typedef bf16  v8bf  __attribute__((ext_vector_type(8)));
typedef float v8f   __attribute__((ext_vector_type(8)));
typedef unsigned int u32x4 __attribute__((ext_vector_type(4)));
typedef int          i32x4 __attribute__((ext_vector_type(4)));
typedef int          i32x8 __attribute__((ext_vector_type(8)));

#define MEL   128
#define TLEN  201
#define TPAD  208     // h row stride (13 tiles of 16)
#define SPAD  224     // s stride for att (7 K-chunks of 32)
#define XLEN  102400
#define XPADL 103424  // 512 + 102400 + 512
#define XWIN  8704    // 15*512 + 1024: x window per 16-t block

#if defined(__has_builtin)
#if __has_builtin(__builtin_amdgcn_tensor_load_to_lds)
#define HAVE_TDM 1
#endif
#endif

#if __has_include(<hip/amd_detail/amd_gfx1250_TDM.h>)
#define TDM_6ARG 1
#endif

// A fragment (16x32 bf16): rows m (k-contiguous storage, leading dim ld).
__device__ __forceinline__ v16bf frag_a(const bf16* base, int ld, int lane) {
  int m  = lane & 15;
  int h8 = (lane >> 4) * 8;
  const bf16* p = base + m * ld + h8;
  v8bf lo = *(const v8bf*)(p);
  v8bf hi = *(const v8bf*)(p + 16);
  return __builtin_shufflevector(lo, hi, 0,1,2,3,4,5,6,7,8,9,10,11,12,13,14,15);
}

// B fragment (32x16 bf16): value(k,n) stored at row n, k-contiguous (ld = row stride).
__device__ __forceinline__ v16bf frag_b(const bf16* base, int ld, int lane) {
  int n   = lane & 15;
  int h8  = (lane >> 4) * 8;
  const bf16* p = base + n * ld + h8;
  v8bf lo = *(const v8bf*)(p);
  v8bf hi = *(const v8bf*)(p + 16);
  return __builtin_shufflevector(lo, hi, 0,1,2,3,4,5,6,7,8,9,10,11,12,13,14,15);
}

__device__ __forceinline__ v8f wmma_bf16(v16bf a, v16bf b, v8f c) {
  return __builtin_amdgcn_wmma_f32_16x16x32_bf16(false, a, false, b, (short)0, c,
                                                 false, false);
}

#ifdef HAVE_TDM
// 1D tile load, global -> LDS, via TDM descriptor (ISA 08_async_tensor §8).
// nelem = #2-byte elements (fits tile_dim0 16 bits). Uniform args required.
__device__ __forceinline__ void tdm_load_1d(unsigned lds_off,
                                            unsigned long long gaddr,
                                            unsigned nelem) {
  u32x4 g0;
  g0[0] = 1u;                                             // count=1, user mode
  g0[1] = lds_off;                                        // lds_addr (bytes)
  g0[2] = (unsigned)(gaddr & 0xFFFFFFFFu);                // global_addr[31:0]
  g0[3] = (unsigned)((gaddr >> 32) & 0x1FFFFFFu) | (2u << 30);  // ga[56:32]|type=2
  i32x8 g1;
  g1[0] = (int)(1u << 16);               // wg_mask=0, data_size=1 (2 bytes)
  g1[1] = (int)(0xFFFFu << 16);          // tensor_dim0[15:0]=0xFFFF (huge, no OOB)
  g1[2] = (int)(0x7FFFu | (1u << 16));   // tensor_dim0[31:16]; tensor_dim1=1
  g1[3] = (int)(nelem << 16);            // tensor_dim1 hi=0; tile_dim0=nelem
  g1[4] = 1;                             // tile_dim1=1, tile_dim2=0
  g1[5] = (int)nelem;                    // tensor_dim0_stride lo
  g1[6] = 0;
  g1[7] = 0;
  i32x4 gz = {0, 0, 0, 0};
#ifdef TDM_6ARG
  i32x8 gz8 = {0, 0, 0, 0, 0, 0, 0, 0};
  __builtin_amdgcn_tensor_load_to_lds(g0, g1, gz, gz, gz8, 0);
#else
  __builtin_amdgcn_tensor_load_to_lds(g0, g1, gz, gz, 0);
#endif
}
#endif

// ---------------------------------------------------------------------------
// Kernel 0: pack / convert weights & padded input to bf16 in workspace.
// ---------------------------------------------------------------------------
__global__ void pack_kernel(const float* __restrict__ x,
                            const float* __restrict__ wext,
                            const float* __restrict__ convw,
                            const float* __restrict__ wq,
                            const float* __restrict__ wk,
                            const float* __restrict__ wv,
                            bf16* __restrict__ xpad,
                            bf16* __restrict__ wextb,
                            bf16* __restrict__ wconvb,
                            bf16* __restrict__ wqb,
                            bf16* __restrict__ wkb,
                            bf16* __restrict__ wvb,
                            bf16* __restrict__ vbG) {
  size_t stride = (size_t)gridDim.x * blockDim.x;
  size_t tid = (size_t)blockIdx.x * blockDim.x + threadIdx.x;
  for (size_t i = tid; i < (size_t)256 * XPADL; i += stride) {
    size_t b = i / XPADL, p = i % XPADL;
    float v = (p >= 512 && p < 512 + XLEN) ? x[b * XLEN + (p - 512)] : 0.f;
    xpad[i] = (bf16)v;
  }
  for (size_t i = tid; i < (size_t)MEL * 1024; i += stride) wextb[i] = (bf16)wext[i];
  // conv weights repacked: k = dt*128 + c (chunk-of-32 has fixed dt, contiguous c)
  for (size_t i = tid; i < (size_t)3 * MEL * 384; i += stride) {
    size_t l = i / (MEL * 384), rem = i % (MEL * 384);
    size_t o = rem / 384, k = rem % 384;
    size_t dt = k >> 7, c = k & 127;
    wconvb[i] = (bf16)convw[((l * MEL + o) * MEL + c) * 3 + dt];
  }
  for (size_t i = tid; i < 16 * 128; i += stride) { wqb[i] = (bf16)wq[i]; wkb[i] = (bf16)wk[i]; }
  for (size_t i = tid; i < 128 * 128; i += stride) wvb[i] = (bf16)wv[i];
  for (size_t i = tid; i < (size_t)256 * MEL * SPAD; i += stride) vbG[i] = (bf16)0.f;
}

// ---------------------------------------------------------------------------
// Kernel 1: extractor GEMM  h0[b,o,t] = sum_k Wext[o,k] * xpad[b, t*512+k]
// grid = (13 t-tiles, 256 batches), 8 waves = 8 M-tiles. The shared 17KB x
// window is staged once per block into LDS with the TDM; B fragments then
// come from LDS while A (weights) streams from L2.
// ---------------------------------------------------------------------------
__global__ void extractor_kernel(const bf16* __restrict__ wext,
                                 const bf16* __restrict__ xpad,
                                 float* __restrict__ hout) {
  __shared__ __align__(32) bf16 xs[XWIN];
  int tt = blockIdx.x, b = blockIdx.y;
  int lane = threadIdx.x & 31, wid = threadIdx.x >> 5;
  int t0 = tt * 16, m0 = wid * 16;
  const bf16* xsrc = xpad + (size_t)b * XPADL + (size_t)t0 * 512;
#ifdef HAVE_TDM
  if (wid == 0) {  // wave-uniform branch: only wave 0 issues the DMA
    unsigned lds_off = (unsigned)(uintptr_t)(void*)&xs[0];
    unsigned long long ga = (unsigned long long)(uintptr_t)xsrc;
    tdm_load_1d(lds_off, ga, XWIN);
    __builtin_amdgcn_s_wait_tensorcnt(0);
  }
#else
  for (int i = threadIdx.x; i < XWIN; i += 256) xs[i] = xsrc[i];
#endif
  __syncthreads();

  v8f acc = {0.f, 0.f, 0.f, 0.f, 0.f, 0.f, 0.f, 0.f};
  for (int ks = 0; ks < 32; ++ks) {
    v16bf a = frag_a(wext + m0 * 1024 + ks * 32, 1024, lane);
    v16bf bb = frag_b(xs + ks * 32, 512, lane);  // window stride 512, k-contiguous
    acc = wmma_bf16(a, bb, acc);
  }
  int n = lane & 15, half = lane >> 4;
  int t = t0 + n;
  if (t < TLEN) {
    float* out = hout + (size_t)b * MEL * TPAD;
#pragma unroll
    for (int r = 0; r < 8; ++r) {
      int m = m0 + r + half * 8;
      out[(size_t)m * TPAD + t] = acc[r];
    }
  }
}

// ---------------------------------------------------------------------------
// Kernel 2: fused LayerNorm(T) + LeakyReLU + Conv1d(128,128,3,pad=1) per batch.
// ---------------------------------------------------------------------------
__global__ void layer_kernel(const float* __restrict__ hin,
                             float* __restrict__ hout,
                             const bf16* __restrict__ wconv,
                             const float* __restrict__ lnw,
                             const float* __restrict__ lnb) {
  __shared__ float mu_s[MEL];
  __shared__ float rs_s[MEL];
  __shared__ __align__(32) bf16 nb[212 * MEL];  // t-major rows with +1 halo
  int b = blockIdx.x;
  int lane = threadIdx.x & 31, wid = threadIdx.x >> 5;
  const float* hb_ = hin + (size_t)b * MEL * TPAD;

  for (int c = wid * 16; c < wid * 16 + 16; ++c) {
    const float* row = hb_ + (size_t)c * TPAD;
    float s1 = 0.f, s2 = 0.f;
    for (int t = lane; t < TLEN; t += 32) { float v = row[t]; s1 += v; s2 += v * v; }
    for (int off = 16; off; off >>= 1) {
      s1 += __shfl_xor(s1, off, 32);
      s2 += __shfl_xor(s2, off, 32);
    }
    if (lane == 0) {
      float mu = s1 * (1.f / TLEN);
      float var = s2 * (1.f / TLEN) - mu * mu;
      mu_s[c] = mu;
      rs_s[c] = rsqrtf(var + 1e-5f);
    }
  }
  __syncthreads();

  for (int idx = threadIdx.x; idx < 212 * MEL; idx += 256) {
    int ts = idx >> 7, c = idx & 127;
    int t = ts - 1;
    float v = 0.f;
    if (t >= 0 && t < TLEN) {
      float xv = hb_[(size_t)c * TPAD + t];
      v = (xv - mu_s[c]) * rs_s[c] * lnw[t] + lnb[t];
      v = v >= 0.f ? v : 0.2f * v;
    }
    nb[idx] = (bf16)v;
  }
  __syncthreads();

  int m0 = wid * 16;
  int n = lane & 15, half = lane >> 4;
  float* outb = hout + (size_t)b * MEL * TPAD;
  for (int tt = 0; tt < 13; ++tt) {
    int t0 = tt * 16;
    v8f acc = {0.f, 0.f, 0.f, 0.f, 0.f, 0.f, 0.f, 0.f};
    for (int ks = 0; ks < 12; ++ks) {
      int dt = ks >> 2, c0 = (ks & 3) * 32;
      v16bf a = frag_a(wconv + m0 * 384 + ks * 32, 384, lane);
      v16bf bb = frag_b(&nb[(t0 + dt) * MEL + c0], MEL, lane);
      acc = wmma_bf16(a, bb, acc);
    }
    int t = t0 + n;
    if (t < TLEN) {
#pragma unroll
      for (int r = 0; r < 8; ++r) {
        int m = m0 + r + half * 8;
        outb[(size_t)m * TPAD + t] = acc[r];
      }
    }
  }
}

// ---------------------------------------------------------------------------
// Kernel 3: attention + residual + GAP + FC, one block per batch.
// ---------------------------------------------------------------------------
__global__ void attn_kernel(const float* __restrict__ hin,
                            const bf16* __restrict__ wqb,
                            const bf16* __restrict__ wkb,
                            const bf16* __restrict__ wvb,
                            const float* __restrict__ bq,
                            const float* __restrict__ bk,
                            const float* __restrict__ bv,
                            const float* __restrict__ gammap,
                            const float* __restrict__ fcw,
                            const float* __restrict__ fcb,
                            bf16* __restrict__ vbG,
                            bf16* __restrict__ qbG,
                            bf16* __restrict__ kbG,
                            float* __restrict__ outG) {
  __shared__ __align__(32) unsigned char smem[55296];
  bf16* hb = (bf16*)smem;                         // [208][128] bf16 (phase 0/1)
  float* etile = (float*)smem;                    // [16][208] f32 (phase 2 overlay)
  bf16* abuf = (bf16*)(smem + 16 * TPAD * 4);     // [16][224] bf16
  float* gacc = (float*)(smem + 53248);           // [128]

  int b = blockIdx.x;
  int lane = threadIdx.x & 31, wid = threadIdx.x >> 5;
  int n = lane & 15, half = lane >> 4;
  int m0 = wid * 16;
  float gamma = gammap[0];
  const float* hsrc = hin + (size_t)b * MEL * TPAD;

  for (int idx = threadIdx.x; idx < TPAD * MEL; idx += 256) {
    int c = idx / TPAD, t = idx % TPAD;
    float v = (t < TLEN) ? hsrc[(size_t)c * TPAD + t] : 0.f;
    hb[t * MEL + c] = (bf16)v;
  }
  if (threadIdx.x < MEL) gacc[threadIdx.x] = 0.f;
  __syncthreads();

  bf16* vrow = vbG + (size_t)b * MEL * SPAD;
  for (int tt = 0; tt < 13; ++tt) {
    v8f acc = {0.f, 0.f, 0.f, 0.f, 0.f, 0.f, 0.f, 0.f};
    for (int ks = 0; ks < 4; ++ks) {
      v16bf a = frag_a(wvb + m0 * MEL + ks * 32, MEL, lane);
      v16bf bb = frag_b(hb + tt * 16 * MEL + ks * 32, MEL, lane);
      acc = wmma_bf16(a, bb, acc);
    }
    int s = tt * 16 + n;
#pragma unroll
    for (int r = 0; r < 8; ++r) {
      int m = m0 + r + half * 8;
      vrow[(size_t)m * SPAD + s] = (bf16)(acc[r] + bv[m]);
    }
  }
  if (wid < 2) {
    const bf16* W = wid ? wkb : wqb;
    const float* bias = wid ? bk : bq;
    bf16* dst = (wid ? kbG : qbG) + (size_t)b * TPAD * 32;
    for (int tt = 0; tt < 13; ++tt) {
      v8f acc = {0.f, 0.f, 0.f, 0.f, 0.f, 0.f, 0.f, 0.f};
      for (int ks = 0; ks < 4; ++ks) {
        v16bf a = frag_a(W + ks * 32, MEL, lane);
        v16bf bb = frag_b(hb + tt * 16 * MEL + ks * 32, MEL, lane);
        acc = wmma_bf16(a, bb, acc);
      }
      int t = tt * 16 + n;
#pragma unroll
      for (int r = 0; r < 8; ++r) {
        int o = r + half * 8;
        float v = (t < TLEN) ? (acc[r] + bias[o]) : 0.f;  // zero invalid rows
        dst[(size_t)t * 32 + o] = (bf16)v;
        dst[(size_t)t * 32 + o + 16] = (bf16)0.f;         // K-pad to 32
      }
    }
  }
  __threadfence();
  __syncthreads();

  const bf16* qrow = qbG + (size_t)b * TPAD * 32;
  const bf16* krow = kbG + (size_t)b * TPAD * 32;
  for (int tt = 0; tt < 13; ++tt) {
    __syncthreads();
    for (int st = wid; st < 13; st += 8) {
      v8f acc = {0.f, 0.f, 0.f, 0.f, 0.f, 0.f, 0.f, 0.f};
      v16bf a = frag_a(qrow + tt * 16 * 32, 32, lane);
      v16bf bb = frag_b(krow + st * 16 * 32, 32, lane);
      acc = wmma_bf16(a, bb, acc);
#pragma unroll
      for (int r = 0; r < 8; ++r) {
        int tr = r + half * 8, sc = st * 16 + n;
        etile[tr * TPAD + sc] = acc[r];
      }
    }
    __syncthreads();
    for (int rr = wid; rr < 16; rr += 8) {
      int t = tt * 16 + rr;
      if (t < TLEN) {
        float mx = -3.0e38f;
        for (int s = lane; s < TLEN; s += 32) mx = fmaxf(mx, etile[rr * TPAD + s]);
        for (int off = 16; off; off >>= 1) mx = fmaxf(mx, __shfl_xor(mx, off, 32));
        float sum = 0.f;
        for (int s = lane; s < TLEN; s += 32) sum += __expf(etile[rr * TPAD + s] - mx);
        for (int off = 16; off; off >>= 1) sum += __shfl_xor(sum, off, 32);
        float inv = 1.f / sum;
        for (int s = lane; s < SPAD; s += 32) {
          float v = (s < TLEN) ? __expf(etile[rr * TPAD + s] - mx) * inv : 0.f;
          abuf[rr * SPAD + s] = (bf16)v;
        }
      } else {
        for (int s = lane; s < SPAD; s += 32) abuf[rr * SPAD + s] = (bf16)0.f;
      }
    }
    __syncthreads();
    {
      v8f acc = {0.f, 0.f, 0.f, 0.f, 0.f, 0.f, 0.f, 0.f};
      for (int ks = 0; ks < 7; ++ks) {
        v16bf a = frag_a((const bf16*)vrow + m0 * SPAD + ks * 32, SPAD, lane);
        v16bf bb = frag_b(abuf + ks * 32, SPAD, lane);
        acc = wmma_bf16(a, bb, acc);
      }
      int t = tt * 16 + n;
      if (t < TLEN) {
#pragma unroll
        for (int r = 0; r < 8; ++r) {
          int c = m0 + r + half * 8;
          atomicAdd(&gacc[c], gamma * acc[r]);
        }
      }
    }
  }
  __syncthreads();

  for (int c = threadIdx.x; c < MEL; c += 256) {
    float s = 0.f;
    const float* row = hsrc + (size_t)c * TPAD;
    for (int t = 0; t < TLEN; ++t) s += row[t];
    gacc[c] += s;
  }
  __syncthreads();

  if (threadIdx.x < MEL) {
    int j = threadIdx.x;
    float s = 0.f;
    for (int c = 0; c < MEL; ++c) s += fcw[j * MEL + c] * gacc[c];
    outG[(size_t)b * MEL + j] = fcb[j] + s * (1.f / TLEN);
  }
}

// ---------------------------------------------------------------------------
// Host launcher
// ---------------------------------------------------------------------------
extern "C" void kernel_launch(void* const* d_in, const int* in_sizes, int n_in,
                              void* d_out, int out_size, void* d_ws, size_t ws_size,
                              hipStream_t stream) {
  const float* x     = (const float*)d_in[0];
  const float* wext  = (const float*)d_in[1];
  const float* lnw   = (const float*)d_in[2];
  const float* lnb   = (const float*)d_in[3];
  const float* convw = (const float*)d_in[4];
  const float* wq    = (const float*)d_in[5];
  const float* bq    = (const float*)d_in[6];
  const float* wk    = (const float*)d_in[7];
  const float* bk    = (const float*)d_in[8];
  const float* wv    = (const float*)d_in[9];
  const float* bv    = (const float*)d_in[10];
  const float* gamma = (const float*)d_in[11];
  const float* fcw   = (const float*)d_in[12];
  const float* fcb   = (const float*)d_in[13];
  float* out = (float*)d_out;
  char* ws = (char*)d_ws;

  constexpr size_t O_XPAD  = 0;
  constexpr size_t O_WEXT  = O_XPAD  + (size_t)256 * XPADL * 2;
  constexpr size_t O_WCONV = O_WEXT  + (size_t)MEL * 1024 * 2;
  constexpr size_t O_WQ    = O_WCONV + (size_t)3 * MEL * 384 * 2;
  constexpr size_t O_WK    = O_WQ    + (size_t)16 * MEL * 2;
  constexpr size_t O_WV    = O_WK    + (size_t)16 * MEL * 2;
  constexpr size_t O_HA    = O_WV    + (size_t)MEL * MEL * 2;
  constexpr size_t O_HB    = O_HA    + (size_t)256 * MEL * TPAD * 4;
  constexpr size_t O_VB    = O_HB    + (size_t)256 * MEL * TPAD * 4;
  constexpr size_t O_QB    = O_VB    + (size_t)256 * MEL * SPAD * 2;
  constexpr size_t O_KB    = O_QB    + (size_t)256 * TPAD * 32 * 2;

  bf16* xpad   = (bf16*)(ws + O_XPAD);
  bf16* wextb  = (bf16*)(ws + O_WEXT);
  bf16* wconvb = (bf16*)(ws + O_WCONV);
  bf16* wqb    = (bf16*)(ws + O_WQ);
  bf16* wkb    = (bf16*)(ws + O_WK);
  bf16* wvb    = (bf16*)(ws + O_WV);
  float* hA    = (float*)(ws + O_HA);
  float* hB    = (float*)(ws + O_HB);
  bf16* vb     = (bf16*)(ws + O_VB);
  bf16* qb     = (bf16*)(ws + O_QB);
  bf16* kb     = (bf16*)(ws + O_KB);

  pack_kernel<<<2048, 256, 0, stream>>>(x, wext, convw, wq, wk, wv,
                                        xpad, wextb, wconvb, wqb, wkb, wvb, vb);
  extractor_kernel<<<dim3(13, 256), 256, 0, stream>>>(wextb, xpad, hA);
  layer_kernel<<<256, 256, 0, stream>>>(hA, hB, wconvb + 0 * MEL * 384, lnw + 0 * TLEN, lnb + 0 * TLEN);
  layer_kernel<<<256, 256, 0, stream>>>(hB, hA, wconvb + 1 * MEL * 384, lnw + 1 * TLEN, lnb + 1 * TLEN);
  layer_kernel<<<256, 256, 0, stream>>>(hA, hB, wconvb + 2 * MEL * 384, lnw + 2 * TLEN, lnb + 2 * TLEN);
  attn_kernel<<<256, 256, 0, stream>>>(hB, wqb, wkb, wvb, bq, bk, bv, gamma,
                                       fcw, fcb, vb, qb, kb, out);
}
